// ATOCActorNet_73177652789591
// MI455X (gfx1250) — matrix operand order repaired and checked by reading, not compiled, not run on hardware
//
#include <hip/hip_runtime.h>
#include <hip/hip_bf16.h>

// ---------------------------------------------------------------------------
// ATOC actor network on MI455X (gfx1250): f16 WMMA (f32 accumulate) for all
// GEMMs, LDS-resident sequential bi-LSTM communication pass.
// ---------------------------------------------------------------------------

typedef __attribute__((ext_vector_type(16))) _Float16 v16h;
typedef __attribute__((ext_vector_type(8)))  float    v8f;

union V16H { v16h v; _Float16 h[16]; unsigned int u[8]; };

static constexpr int BB    = 128;   // batch
static constexpr int NN    = 128;   // agents
static constexpr int TT    = 256;   // thought dim
static constexpr int OBSD  = 512;
static constexpr int ACTD  = 64;
static constexpr int MMG   = 16;    // group size
static constexpr int EMBD  = 128;
static constexpr int HHD   = 128;   // LSTM hidden per direction (4H = 512)
static constexpr int RR    = BB * NN;  // 16384 rows

// ---------------- WMMA fragment helpers (wave32; A/B in contiguous k = 16*(lane/16) + s order) --------

// A fragment: 16x32 f16, row-major source with leading dim ld.
__device__ __forceinline__ v16h load_a_rm(const _Float16* __restrict__ src,
                                          int ld, int row0, int k0, int lane) {
  const int row = row0 + (lane & 15);
  const int kh  = (lane >> 4) << 4;          // lanes 16-31 hold K+16 halves
  V16H r;
#pragma unroll
  for (int v = 0; v < 8; ++v) {
    const int kb = k0 + kh + (v << 1);
    r.u[v] = *reinterpret_cast<const unsigned int*>(src + (long)row * ld + kb);
  }
  return r.v;
}

// A fragment from LDS group buffer (16 rows x 256), optional row reversal.
__device__ __forceinline__ v16h load_a_lds(const _Float16* __restrict__ src,
                                           int k0, int lane, int rev) {
  int row = lane & 15;
  if (rev) row = 15 - row;
  const int kh = (lane >> 4) << 4;
  V16H r;
#pragma unroll
  for (int v = 0; v < 8; ++v) {
    const int kb = k0 + kh + (v << 1);
    r.u[v] = *reinterpret_cast<const unsigned int*>(src + row * TT + kb);
  }
  return r.v;
}

// B fragment: 32x16 f16 from row-major [K, N] weight, leading dim ld == N.
__device__ __forceinline__ v16h load_b_rm(const _Float16* __restrict__ W,
                                          int ld, int k0, int n0, int lane) {
  const int n  = n0 + (lane & 15);
  const int ko = k0 + ((lane >> 4) << 4);    // lanes 16-31 hold K+16 rows
  V16H r;
#pragma unroll
  for (int v = 0; v < 8; ++v) {
    r.h[2 * v]     = W[(long)(ko + 2 * v)     * ld + n];
    r.h[2 * v + 1] = W[(long)(ko + 2 * v + 1) * ld + n];
  }
  return r.v;
}

// B fragment where B(k,n) = X[n][k] (i.e. X^T), X row-major leading dim ld.
__device__ __forceinline__ v16h load_bt_rm(const _Float16* __restrict__ X,
                                           int ld, int k0, int n0, int lane) {
  const int n  = n0 + (lane & 15);
  const int ko = k0 + ((lane >> 4) << 4);
  V16H r;
#pragma unroll
  for (int v = 0; v < 8; ++v)
    r.u[v] = *reinterpret_cast<const unsigned int*>(X + (long)n * ld + ko + 2 * v);
  return r.v;
}

// C/D fragment store: lanes 0-15 col n, rows 0-7; lanes 16-31 rows 8-15.
__device__ __forceinline__ void store_c_rm(float* __restrict__ dst, int ld,
                                           int row0, int col0, v8f c, int lane) {
  const int col = col0 + (lane & 15);
  const int rb  = row0 + ((lane >> 4) << 3);
#pragma unroll
  for (int r = 0; r < 8; ++r) dst[(long)(rb + r) * ld + col] = c[r];
}

__device__ __forceinline__ v8f wmma16(v16h a, v16h b, v8f c) {
  return __builtin_amdgcn_wmma_f32_16x16x32_f16(
      /*neg_a=*/false, a, /*neg_b=*/false, b,
      /*c_mod=*/(short)0, c, /*reuse_a=*/false, /*reuse_b=*/false);
}

__device__ __forceinline__ float sigf(float x) { return 1.f / (1.f + __expf(-x)); }

// ---------------- fused GEMM (+bias) [+LayerNorm] [+act] --------------------
// Block owns 16 rows x NC cols (full rows => LN fits in-block).
// NC/16 waves, each wave one 16-wide column tile; K-loop of 16x16x32 WMMAs.
template <int NC, bool DO_LN, int ACTM>   // ACTM: 0 none, 1 relu, 2 tanh
__global__ void __launch_bounds__(2 * NC)
gemm16_kernel(const _Float16* __restrict__ X, int K,
              const _Float16* __restrict__ W,   // [K, NC] row-major, f16
              const float* __restrict__ bias,
              const float* __restrict__ gamma,
              const float* __restrict__ beta,
              float* __restrict__ outf,
              float* __restrict__ outf2,
              _Float16* __restrict__ outh) {
  __shared__ float tile[16][NC + 1];
  const int row0 = blockIdx.x * 16;
  const int wave = threadIdx.x >> 5;
  const int lane = threadIdx.x & 31;
  const int n0   = wave * 16;

  v8f c = {};
  for (int k0 = 0; k0 < K; k0 += 32) {
    v16h a = load_a_rm(X, K, row0, k0, lane);
    v16h b = load_b_rm(W, NC, k0, n0, lane);
    c = wmma16(a, b, c);
  }
  {
    const int col = n0 + (lane & 15);
    const int rb  = (lane >> 4) << 3;
    const float bv = bias[col];
#pragma unroll
    for (int r = 0; r < 8; ++r) tile[rb + r][col] = c[r] + bv;
  }
  __syncthreads();

  const int NW = NC / 16;                 // waves in block
  for (int r = wave; r < 16; r += NW) {
    float mu = 0.f, inv = 1.f;
    if (DO_LN) {
      float s = 0.f, s2 = 0.f;
      for (int c0 = lane; c0 < NC; c0 += 32) {
        const float x = tile[r][c0];
        s += x; s2 += x * x;
      }
#pragma unroll
      for (int off = 16; off > 0; off >>= 1) {
        s  += __shfl_xor(s,  off, 32);
        s2 += __shfl_xor(s2, off, 32);
      }
      mu = s / NC;
      const float var = s2 / NC - mu * mu;
      inv = rsqrtf(var + 1e-5f);
    }
    for (int c0 = lane; c0 < NC; c0 += 32) {
      float y = tile[r][c0];
      if (DO_LN) y = (y - mu) * inv * gamma[c0] + beta[c0];
      if (ACTM == 1) y = fmaxf(y, 0.f);
      if (ACTM == 2) y = tanhf(y);
      const long o = (long)(row0 + r) * NC + c0;
      if (outf)  outf[o]  = y;
      if (outf2) outf2[o] = y;
      if (outh)  outh[o]  = (_Float16)y;
    }
  }
}

// ---------------- attention head: p = sigmoid(a2 @ Wa3 + ba3) ---------------
__global__ void att_final_kernel(const _Float16* __restrict__ a2,
                                 const float* __restrict__ Wa3,
                                 const float* __restrict__ ba3,
                                 float* __restrict__ prob,
                                 float* __restrict__ flag_out,
                                 float* __restrict__ flags_ws) {
  const int i = blockIdx.x * blockDim.x + threadIdx.x;
  if (i >= RR) return;
  float s = ba3[0];
  const _Float16* a = a2 + (long)i * EMBD;
  for (int k = 0; k < EMBD; ++k) s += (float)a[k] * Wa3[k];
  const float p  = sigf(s);
  const float fl = (p > 0.4f) ? 1.f : 0.f;
  prob[i] = p; flag_out[i] = fl; flags_ws[i] = fl;
}

// ---------------- pairwise dot: dot[b] = X X^T  (WMMA) ----------------------
__global__ void __launch_bounds__(512)
dot_kernel(const _Float16* __restrict__ th, float* __restrict__ dotw) {
  const int b  = blockIdx.x;
  const int rb = blockIdx.y;                // 4 row-blocks of 32 rows
  const _Float16* X = th + (long)b * NN * TT;
  const int wave = threadIdx.x >> 5;
  const int lane = threadIdx.x & 31;
  const int r0 = rb * 32 + ((wave >> 3) << 4);
  const int n0 = (wave & 7) << 4;
  v8f c = {};
  for (int k0 = 0; k0 < TT; k0 += 32) {
    v16h a  = load_a_rm(X, TT, r0, k0, lane);
    v16h bt = load_bt_rm(X, TT, k0, n0, lane);
    c = wmma16(a, bt, c);
  }
  store_c_rm(dotw + (long)b * NN * NN, NN, r0, n0, c, lane);
}

// ---------------- top-16 nearest + adjacency C ------------------------------
__global__ void topk_kernel(const float* __restrict__ dotw,
                            const float* __restrict__ flags,
                            int* __restrict__ idx_out,
                            float* __restrict__ C_out) {
  const int gid = blockIdx.x * blockDim.x + threadIdx.x;
  if (gid >= RR) return;
  const int b = gid >> 7, r = gid & 127;
  const float* db   = dotw + (long)b * NN * NN;
  const float* drow = db + (long)r * NN;
  const float  dr   = db[(long)r * NN + r];
  unsigned mask[4] = {0u, 0u, 0u, 0u};
  int sel[MMG];
  for (int m = 0; m < MMG; ++m) {
    float best = 3.4e38f; int bi = 0;
    for (int cc = 0; cc < NN; ++cc) {
      if ((mask[cc >> 5] >> (cc & 31)) & 1u) continue;
      const float dist = dr - 2.f * drow[cc] + db[(long)cc * NN + cc];
      if (dist < best) { best = dist; bi = cc; }  // strict <: stable tie-break
    }
    sel[m] = bi;
    mask[bi >> 5] |= 1u << (bi & 31);
  }
  for (int a = 1; a < MMG; ++a) {                  // ascending (jnp.sort)
    const int v = sel[a]; int j = a - 1;
    while (j >= 0 && sel[j] > v) { sel[j + 1] = sel[j]; --j; }
    sel[j + 1] = v;
  }
  int* io = idx_out + (long)gid * MMG;
  for (int m = 0; m < MMG; ++m) io[m] = sel[m];
  const float fl = flags[gid];
  float* Crow = C_out + (long)b * NN * NN + (long)r * NN;
  for (int cc = 0; cc < NN; ++cc) Crow[cc] = 0.f;
  for (int m = 0; m < MMG; ++m) Crow[sel[m]] = fl;
}

// ---------------- sequential bi-LSTM communication --------------------------
// One block (1024 thr) per batch; whole per-batch thoughts matrix in LDS.
__global__ void __launch_bounds__(1024)
comm_kernel(float* __restrict__ newT,
            const float* __restrict__ flags,
            const int* __restrict__ idx,
            const _Float16* __restrict__ WfihT, const _Float16* __restrict__ WfhhT,
            const _Float16* __restrict__ WbihT, const _Float16* __restrict__ WbhhT,
            const float* __restrict__ bf_ih, const float* __restrict__ bf_hh,
            const float* __restrict__ bb_ih, const float* __restrict__ bb_hh) {
  extern __shared__ char smem[];
  float*    thb   = (float*)smem;                      // [128][256]
  _Float16* grpH  = (_Float16*)(thb + NN * TT);        // [16][256] f16
  float*    zin   = (float*)(grpH + MMG * TT);         // [2][16][512]
  float*    gates = zin + 2 * MMG * 512;               // [2][512]
  float*    hbuf  = gates + 1024;                      // [2][16][128]
  float*    cbuf  = hbuf + 2 * MMG * HHD;              // [2][128]

  const int b   = blockIdx.x;
  float* Tb     = newT + (long)b * NN * TT;
  const int tid = threadIdx.x;
  const int wave = tid >> 5;
  const int lane = tid & 31;

  for (int e = tid; e < NN * TT; e += 1024) thb[e] = Tb[e];
  __syncthreads();

  for (int agent = 0; agent < NN; ++agent) {
    const float fl = flags[b * NN + agent];
    if (fl == 0.f) continue;                 // block-uniform
    const int* gi = idx + ((long)b * NN + agent) * MMG;

    // gather group rows -> f16 (for WMMA A)
    for (int e = tid; e < MMG * TT; e += 1024) {
      const int m = e >> 8, k = e & 255;
      grpH[m * TT + k] = (_Float16)thb[gi[m] * TT + k];
    }
    if (tid < 2 * HHD) cbuf[tid] = 0.f;
    __syncthreads();

    // input projections: zin[dir] = grp(_rev) @ WihT + (bih + bhh); 64 tiles / 32 waves
#pragma unroll
    for (int t2 = 0; t2 < 2; ++t2) {
      const int tileid = wave * 2 + t2;       // 0..63
      const int dir    = tileid >> 5;         // 0 fwd, 1 bwd (reversed rows)
      const int n0     = (tileid & 31) * 16;
      const _Float16* WT = dir ? WbihT : WfihT;   // [256][512]
      const float* bi1 = dir ? bb_ih : bf_ih;
      const float* bi2 = dir ? bb_hh : bf_hh;
      const int col = n0 + (lane & 15);
      const float bv = bi1[col] + bi2[col];
      v8f c;
#pragma unroll
      for (int r = 0; r < 8; ++r) c[r] = bv;
      for (int k0 = 0; k0 < TT; k0 += 32) {
        v16h a   = load_a_lds(grpH, k0, lane, dir);
        v16h bfr = load_b_rm(WT, 4 * HHD, k0, n0, lane);
        c = wmma16(a, bfr, c);
      }
      const int rb = (lane >> 4) << 3;
#pragma unroll
      for (int r = 0; r < 8; ++r) zin[(dir * MMG + rb + r) * 512 + col] = c[r];
    }
    __syncthreads();

    // recurrence: 1024 gate dots per step (512 fwd + 512 bwd)
    for (int t = 0; t < MMG; ++t) {
      {
        const int dir = tid >> 9;
        const int n   = tid & 511;
        const _Float16* WT = dir ? WbhhT : WfhhT;   // [128][512]
        float s = zin[(dir * MMG + t) * 512 + n];
        if (t > 0) {
          const float* hp = hbuf + (dir * MMG + (t - 1)) * HHD;
          for (int k = 0; k < HHD; ++k) s += hp[k] * (float)WT[k * 512 + n];
        }
        gates[dir * 512 + n] = s;
      }
      __syncthreads();
      if (tid < 2 * HHD) {
        const int dir = tid >> 7;
        const int j   = tid & 127;
        const float* g = gates + dir * 512;         // torch order i,f,g,o
        const float ig = sigf(g[j]);
        const float fg = sigf(g[128 + j]);
        const float gg = tanhf(g[256 + j]);
        const float og = sigf(g[384 + j]);
        const float cc = fg * cbuf[dir * HHD + j] + ig * gg;
        cbuf[dir * HHD + j] = cc;
        hbuf[(dir * MMG + t) * HHD + j] = og * tanhf(cc);
      }
      __syncthreads();
    }

    // writeback: integ[m] = [hf[m], hb_seq[15-m]]
    for (int e = tid; e < MMG * TT; e += 1024) {
      const int m = e >> 8, k = e & 255;
      const float v = (k < HHD) ? hbuf[m * HHD + k]
                                : hbuf[(MMG + (15 - m)) * HHD + (k - HHD)];
      thb[gi[m] * TT + k] = v;
    }
    __syncthreads();
  }

  for (int e = tid; e < NN * TT; e += 1024) Tb[e] = thb[e];
}

// ---------------- conversion helpers ---------------------------------------
__global__ void cvt_f16(const float* __restrict__ s, _Float16* __restrict__ d, long n) {
  const long i = (long)blockIdx.x * blockDim.x + threadIdx.x;
  if (i < n) d[i] = (_Float16)s[i];
}
__global__ void cvt_f16_relu(const float* __restrict__ s, _Float16* __restrict__ d, long n) {
  const long i = (long)blockIdx.x * blockDim.x + threadIdx.x;
  if (i < n) d[i] = (_Float16)fmaxf(s[i], 0.f);
}
// src [rows, cols] f32 -> dst [cols, rows] f16
__global__ void cvt_f16_T(const float* __restrict__ s, _Float16* __restrict__ d,
                          int rows, int cols) {
  const long i = (long)blockIdx.x * blockDim.x + threadIdx.x;
  if (i < (long)rows * cols) {
    const int r = (int)(i / cols);
    const int c = (int)(i % cols);
    d[(long)c * rows + r] = (_Float16)s[i];
  }
}

// ---------------------------------------------------------------------------
extern "C" void kernel_launch(void* const* d_in, const int* in_sizes, int n_in,
                              void* d_out, int out_size, void* d_ws, size_t ws_size,
                              hipStream_t stream) {
  (void)in_sizes; (void)n_in; (void)out_size; (void)ws_size;
  const float* obs   = (const float*)d_in[0];
  const float* W1    = (const float*)d_in[1];
  const float* b1    = (const float*)d_in[2];
  const float* g1    = (const float*)d_in[3];
  const float* be1   = (const float*)d_in[4];
  const float* W2    = (const float*)d_in[5];
  const float* b2    = (const float*)d_in[6];
  const float* g2    = (const float*)d_in[7];
  const float* be2   = (const float*)d_in[8];
  const float* Wa1   = (const float*)d_in[9];
  const float* ba1   = (const float*)d_in[10];
  const float* Wa2   = (const float*)d_in[11];
  const float* ba2   = (const float*)d_in[12];
  const float* Wa3   = (const float*)d_in[13];
  const float* ba3   = (const float*)d_in[14];
  const float* Wf_ih = (const float*)d_in[15];
  const float* Wf_hh = (const float*)d_in[16];
  const float* bf_ih = (const float*)d_in[17];
  const float* bf_hh = (const float*)d_in[18];
  const float* Wb_ih = (const float*)d_in[19];
  const float* Wb_hh = (const float*)d_in[20];
  const float* bb_ih = (const float*)d_in[21];
  const float* bb_hh = (const float*)d_in[22];
  const float* W3    = (const float*)d_in[23];
  const float* b3    = (const float*)d_in[24];
  const float* g3    = (const float*)d_in[25];
  const float* be3   = (const float*)d_in[26];
  const float* W4    = (const float*)d_in[27];
  const float* b4    = (const float*)d_in[28];
  const float* g4    = (const float*)d_in[29];
  const float* be4   = (const float*)d_in[30];

  // outputs flat in return order
  float* out  = (float*)d_out;
  float* acts = out;                               // [B,N,64]
  float* Cmat = acts + (long)RR * ACTD;            // [B,N,N]
  float* prob = Cmat + (long)BB * NN * NN;         // [B,N,1]
  float* isin = prob + RR;                         // [B,N]
  float* newT = isin + RR;                         // [B,N,T] (mutated by comm)
  float* oldT = newT + (long)RR * TT;              // [B,N,T]

  // workspace carve
  char* w = (char*)d_ws;
  auto take = [&](size_t bytes) -> char* {
    char* p = w; w += (bytes + 255) & ~(size_t)255; return p;
  };
  _Float16* obs_h = (_Float16*)take((size_t)RR * OBSD * 2);
  _Float16* W1h   = (_Float16*)take((size_t)OBSD * TT * 2);
  _Float16* W2h   = (_Float16*)take((size_t)TT * TT * 2);
  _Float16* Wa1h  = (_Float16*)take((size_t)TT * EMBD * 2);
  _Float16* Wa2h  = (_Float16*)take((size_t)EMBD * EMBD * 2);
  _Float16* W3h   = (_Float16*)take((size_t)TT * TT * 2);
  _Float16* W4h   = (_Float16*)take((size_t)TT * ACTD * 2);
  _Float16* WfihT = (_Float16*)take((size_t)TT * 4 * HHD * 2);
  _Float16* WfhhT = (_Float16*)take((size_t)HHD * 4 * HHD * 2);
  _Float16* WbihT = (_Float16*)take((size_t)TT * 4 * HHD * 2);
  _Float16* WbhhT = (_Float16*)take((size_t)HHD * 4 * HHD * 2);
  _Float16* h1h   = (_Float16*)take((size_t)RR * TT * 2);   // also reused for h2 LN out
  _Float16* thh   = (_Float16*)take((size_t)RR * TT * 2);
  _Float16* a1h   = (_Float16*)take((size_t)RR * EMBD * 2);
  _Float16* a2h   = (_Float16*)take((size_t)RR * EMBD * 2);
  _Float16* h2h   = (_Float16*)take((size_t)RR * TT * 2);
  float* dotws    = (float*)take((size_t)BB * NN * NN * 4);
  int*   idxws    = (int*)take((size_t)RR * MMG * 4);
  float* flws     = (float*)take((size_t)RR * 4);

  auto cvt = [&](const float* s, _Float16* d, long n) {
    cvt_f16<<<(int)((n + 255) / 256), 256, 0, stream>>>(s, d, n);
  };
  cvt(obs, obs_h, (long)RR * OBSD);
  cvt(W1,  W1h,  (long)OBSD * TT);
  cvt(W2,  W2h,  (long)TT * TT);
  cvt(Wa1, Wa1h, (long)TT * EMBD);
  cvt(Wa2, Wa2h, (long)EMBD * EMBD);
  cvt(W3,  W3h,  (long)TT * TT);
  cvt(W4,  W4h,  (long)TT * ACTD);
  auto cvtT = [&](const float* s, _Float16* d, int rows, int cols) {
    const long n = (long)rows * cols;
    cvt_f16_T<<<(int)((n + 255) / 256), 256, 0, stream>>>(s, d, rows, cols);
  };
  cvtT(Wf_ih, WfihT, 4 * HHD, TT);    // -> [256,512]: B(k,n)=Wih[n][k]
  cvtT(Wf_hh, WfhhT, 4 * HHD, HHD);   // -> [128,512]
  cvtT(Wb_ih, WbihT, 4 * HHD, TT);
  cvtT(Wb_hh, WbhhT, 4 * HHD, HHD);

  // actor_1: relu(LN(obs@W1+b1)) -> h1h ; thoughts = LN(h1@W2+b2)
  gemm16_kernel<TT, true, 1><<<RR / 16, 2 * TT, 0, stream>>>(
      obs_h, OBSD, W1h, b1, g1, be1, nullptr, nullptr, h1h);
  gemm16_kernel<TT, true, 0><<<RR / 16, 2 * TT, 0, stream>>>(
      h1h, TT, W2h, b2, g2, be2, oldT, newT, thh);

  // attention MLP + sigmoid head
  gemm16_kernel<EMBD, false, 1><<<RR / 16, 2 * EMBD, 0, stream>>>(
      thh, TT, Wa1h, ba1, nullptr, nullptr, nullptr, nullptr, a1h);
  gemm16_kernel<EMBD, false, 1><<<RR / 16, 2 * EMBD, 0, stream>>>(
      a1h, EMBD, Wa2h, ba2, nullptr, nullptr, nullptr, nullptr, a2h);
  att_final_kernel<<<RR / 256, 256, 0, stream>>>(a2h, Wa3, ba3, prob, isin, flws);

  // pairwise distances -> top-16 neighbors -> C
  dot_kernel<<<dim3(BB, 4), 512, 0, stream>>>(thh, dotws);
  topk_kernel<<<RR / 256, 256, 0, stream>>>(dotws, flws, idxws, Cmat);

  // sequential bi-LSTM communication (mutates newT)
  const size_t comm_lds =
      (size_t)(NN * TT + 2 * MMG * 512 + 1024 + 2 * MMG * HHD + 2 * HHD) * 4 +
      (size_t)MMG * TT * 2;   // 226304 B (< 320KB/WGP on CDNA5)
  comm_kernel<<<BB, 1024, comm_lds, stream>>>(
      newT, flws, idxws, WfihT, WfhhT, WbihT, WbhhT, bf_ih, bf_hh, bb_ih, bb_hh);

  // actor_2: relu -> W3+LN -> W4+LN -> tanh
  cvt_f16_relu<<<(int)(((long)RR * TT + 255) / 256), 256, 0, stream>>>(
      newT, h2h, (long)RR * TT);
  gemm16_kernel<TT, true, 0><<<RR / 16, 2 * TT, 0, stream>>>(
      h2h, TT, W3h, b3, g3, be3, nullptr, nullptr, h1h);   // reuse h1h
  gemm16_kernel<ACTD, true, 2><<<RR / 16, 2 * ACTD, 0, stream>>>(
      h1h, TT, W4h, b4, g4, be4, acts, nullptr, nullptr);
}
